// MultiHeadLinearBatchedTokenMixers_75007308857794
// MI455X (gfx1250) — matrix operand order repaired and checked by reading, not compiled, not run on hardware
//
#include <hip/hip_runtime.h>
#include <hip/hip_bf16.h>

typedef __attribute__((ext_vector_type(16))) __bf16       v16bf;
typedef __attribute__((ext_vector_type(8)))  float        v8f;
typedef __attribute__((ext_vector_type(4)))  unsigned int v4u;

#define NEXP  8
#define NH    16
#define NTOK  512   // N (tokens)
#define HD    64    // head dim
#define TOPK  2

union ABvec { v16bf v; unsigned int u[8]; v4u q[2]; };

__device__ __forceinline__ unsigned int pack2_bf16(float a, float b) {
  __bf16 ha = (__bf16)a, hb = (__bf16)b;
  return (unsigned int)__builtin_bit_cast(unsigned short, ha) |
        ((unsigned int)__builtin_bit_cast(unsigned short, hb) << 16);
}

// softmax prob -> (hi, lo) bf16 pair, packed two-at-a-time
__device__ __forceinline__ void split_pair(float w0, float w1, float m, float rinv,
                                           unsigned int& hi, unsigned int& lo) {
  float p0 = __expf(w0 - m) * rinv;
  float p1 = __expf(w1 - m) * rinv;
  __bf16 h0 = (__bf16)p0, h1 = (__bf16)p1;
  float r0 = p0 - (float)h0;
  float r1 = p1 - (float)h1;
  hi = (unsigned int)__builtin_bit_cast(unsigned short, h0) |
      ((unsigned int)__builtin_bit_cast(unsigned short, h1) << 16);
  lo = pack2_bf16(r0, r1);
}

extern "C" __global__ __launch_bounds__(512)
void moe_token_mixer_kernel(const float* __restrict__ x,
                            const long long* __restrict__ expert_indices,
                            const float* __restrict__ expert_weights,
                            const float* __restrict__ weight,
                            const float* __restrict__ bias,
                            float* __restrict__ out)
{
  extern __shared__ unsigned int smem[];         // [2][HD*NTOK/2] packed bf16 pairs
  unsigned int* xh32 = smem;                     // 16384 u32 (64 KB) : x hi
  unsigned int* xl32 = smem + (HD * NTOK / 2);   // 16384 u32 (64 KB) : x lo (residual)

  const int bh    = blockIdx.x;        // b*NH + h
  const int h     = bh & (NH - 1);
  const int tid   = threadIdx.x;
  const int lane  = tid & 31;
  const int wave  = tid >> 5;          // 0..15
  const int row16 = lane & 15;
  const int khalf = lane >> 4;         // 0/1

  // ---- Phase 0: stage x[b,h] (64x512 f32) into LDS as bf16 hi/lo ----
  {
    const float2* xsrc = (const float2*)(x + (size_t)bh * HD * NTOK);
    for (int t = tid; t < HD * NTOK / 2; t += blockDim.x) {
      float2 v = xsrc[t];
      __bf16 h0 = (__bf16)v.x, h1 = (__bf16)v.y;
      xh32[t] = (unsigned int)__builtin_bit_cast(unsigned short, h0) |
               ((unsigned int)__builtin_bit_cast(unsigned short, h1) << 16);
      xl32[t] = pack2_bf16(v.x - (float)h0, v.y - (float)h1);
    }
  }
  __syncthreads();

  const long long eidx[TOPK] = { expert_indices[bh * TOPK + 0], expert_indices[bh * TOPK + 1] };
  const float     egate[TOPK] = { expert_weights[bh * TOPK + 0], expert_weights[bh * TOPK + 1] };

  // each wave handles i-tiles {wave, wave+16}
  for (int itt = 0; itt < 2; ++itt) {
    const int i_tile = wave + itt * 16;
    const int i_row  = i_tile * 16 + row16;            // A-operand row (softmax row i)
    const int i_obase = i_tile * 16 + khalf * 8;       // D rows this lane stores: +r (r=0..7)

    v8f   outacc[4];
    float biasacc[8];
    #pragma unroll
    for (int dt = 0; dt < 4; ++dt) {
      v8f z = {0.f,0.f,0.f,0.f,0.f,0.f,0.f,0.f};
      outacc[dt] = z;
    }
    #pragma unroll
    for (int r = 0; r < 8; ++r) biasacc[r] = 0.f;

    for (int e = 0; e < TOPK; ++e) {
      const long long ei = eidx[e];
      const float     sw = egate[e];
      const float* wrow = weight + (((size_t)ei * NH + h) * NTOK + i_row) * NTOK;

      // ---- softmax row stats: max then sum(exp), each lane does half the row ----
      const float4* wr4 = (const float4*)(wrow + khalf * 256);
      float m = -3.402823466e38f;
      for (int t = 0; t < 64; ++t) {
        float4 w = wr4[t];
        m = fmaxf(m, fmaxf(fmaxf(w.x, w.y), fmaxf(w.z, w.w)));
      }
      m = fmaxf(m, __shfl_xor(m, 16, 32));
      float s = 0.f;
      for (int t = 0; t < 64; ++t) {
        float4 w = wr4[t];
        s += __expf(w.x - m) + __expf(w.y - m) + __expf(w.z - m) + __expf(w.w - m);
      }
      s += __shfl_xor(s, 16, 32);
      const float rinv = 1.0f / s;

      // ---- GEMM: D[i,d] = sum_j P[i,j] * X[d,j], K-chunks of 32, bf16x3 split ----
      v8f acc[4];
      #pragma unroll
      for (int dt = 0; dt < 4; ++dt) {
        v8f z = {0.f,0.f,0.f,0.f,0.f,0.f,0.f,0.f};
        acc[dt] = z;
      }

      for (int kc = 0; kc < 16; ++kc) {
        const int jb = kc * 32 + khalf * 8;
        // A-layout pairs: v<4 -> j = jb+2v ; v>=4 -> j = jb+16+2(v-4)
        float4 wa0 = *(const float4*)(wrow + jb);       // v0,v1
        float4 wa1 = *(const float4*)(wrow + jb + 4);   // v2,v3
        float4 wa2 = *(const float4*)(wrow + jb + 16);  // v4,v5
        float4 wa3 = *(const float4*)(wrow + jb + 20);  // v6,v7

        ABvec Ah, Al;
        split_pair(wa0.x, wa0.y, m, rinv, Ah.u[0], Al.u[0]);
        split_pair(wa0.z, wa0.w, m, rinv, Ah.u[1], Al.u[1]);
        split_pair(wa1.x, wa1.y, m, rinv, Ah.u[2], Al.u[2]);
        split_pair(wa1.z, wa1.w, m, rinv, Ah.u[3], Al.u[3]);
        split_pair(wa2.x, wa2.y, m, rinv, Ah.u[4], Al.u[4]);
        split_pair(wa2.z, wa2.w, m, rinv, Ah.u[5], Al.u[5]);
        split_pair(wa3.x, wa3.y, m, rinv, Ah.u[6], Al.u[6]);
        split_pair(wa3.z, wa3.w, m, rinv, Ah.u[7], Al.u[7]);

        #pragma unroll
        for (int dt = 0; dt < 4; ++dt) {
          const int d    = dt * 16 + row16;                   // B N-column = lane%16
          const int bidx = d * (NTOK / 2) + kc * 16 + khalf * 4;
          ABvec Bh, Bl;
          Bh.q[0] = *(const v4u*)(xh32 + bidx);
          Bh.q[1] = *(const v4u*)(xh32 + bidx + 8);
          Bl.q[0] = *(const v4u*)(xl32 + bidx);
          Bl.q[1] = *(const v4u*)(xl32 + bidx + 8);
          acc[dt] = __builtin_amdgcn_wmma_f32_16x16x32_bf16(false, Ah.v, false, Bh.v,
                                                            (short)0, acc[dt], false, false);
          acc[dt] = __builtin_amdgcn_wmma_f32_16x16x32_bf16(false, Ah.v, false, Bl.v,
                                                            (short)0, acc[dt], false, false);
          acc[dt] = __builtin_amdgcn_wmma_f32_16x16x32_bf16(false, Al.v, false, Bh.v,
                                                            (short)0, acc[dt], false, false);
        }
      }

      // gate-weighted combine over experts
      #pragma unroll
      for (int dt = 0; dt < 4; ++dt)
        #pragma unroll
        for (int r = 0; r < 8; ++r)
          outacc[dt][r] += sw * acc[dt][r];
      #pragma unroll
      for (int r = 0; r < 8; ++r)
        biasacc[r] += sw * bias[((size_t)ei * NH + h) * NTOK + i_obase + r];
    }

    // ---- store: out[b,h,d,i] ; D rows r -> consecutive i, two b128 stores per tile
    float* obase = out + (size_t)bh * HD * NTOK;
    #pragma unroll
    for (int dt = 0; dt < 4; ++dt) {
      const int d = dt * 16 + row16;
      float4 v0, v1;
      v0.x = outacc[dt][0] + biasacc[0]; v0.y = outacc[dt][1] + biasacc[1];
      v0.z = outacc[dt][2] + biasacc[2]; v0.w = outacc[dt][3] + biasacc[3];
      v1.x = outacc[dt][4] + biasacc[4]; v1.y = outacc[dt][5] + biasacc[5];
      v1.z = outacc[dt][6] + biasacc[6]; v1.w = outacc[dt][7] + biasacc[7];
      *(float4*)(obase + (size_t)d * NTOK + i_obase)     = v0;
      *(float4*)(obase + (size_t)d * NTOK + i_obase + 4) = v1;
    }
  }
}

extern "C" void kernel_launch(void* const* d_in, const int* in_sizes, int n_in,
                              void* d_out, int out_size, void* d_ws, size_t ws_size,
                              hipStream_t stream) {
  (void)in_sizes; (void)n_in; (void)out_size; (void)d_ws; (void)ws_size;
  const float*     x   = (const float*)d_in[0];
  const long long* ei  = (const long long*)d_in[1];   // int64 expert indices
  const float*     ew  = (const float*)d_in[2];
  const float*     w   = (const float*)d_in[3];
  const float*     b   = (const float*)d_in[4];
  float*           out = (float*)d_out;

  const int numBH = 8 * NH;                         // B*H = 128 workgroups
  const size_t ldsBytes = 2u * (HD * NTOK / 2) * sizeof(unsigned int);  // 128 KB
  moe_token_mixer_kernel<<<dim3(numBH), dim3(512), ldsBytes, stream>>>(x, ei, ew, w, b, out);
}
